// EvRoomDetector_26929444946615
// MI455X (gfx1250) — matrix operand order repaired and checked by reading, not compiled, not run on hardware
//
#include <hip/hip_runtime.h>
#include <hip/hip_bf16.h>

typedef __attribute__((ext_vector_type(16))) _Float16 v16h;
typedef __attribute__((ext_vector_type(8)))  _Float16 v8h;
typedef __attribute__((ext_vector_type(8)))  float    v8f;
typedef __attribute__((ext_vector_type(4))) unsigned  u32x4;
typedef __attribute__((ext_vector_type(8))) int       i32x8;
typedef __attribute__((ext_vector_type(4))) int       i32x4;

#define H_ 128
#define W_ 128
#define C_ 128
#define NPTS0 32
#define NPTS1 8
#define TOPK 50
#define NLINES (3 * H_ * W_)   /* 49152 */
#define DFEAT 1024
#define HID 1024
#define BK 64                  /* K-step per pipeline stage */

// ---- CDNA5 feature detection -------------------------------------------------
#if defined(__has_builtin)
#  if __has_builtin(__builtin_amdgcn_tensor_load_to_lds)
#    define USE_TDM 1
#  endif
#  if __has_builtin(__builtin_amdgcn_s_wait_tensorcnt)
#    define WAIT_TENSORCNT(n) __builtin_amdgcn_s_wait_tensorcnt(n)
#  endif
#endif
#ifndef USE_TDM
#  define USE_TDM 0
#endif
#ifndef WAIT_TENSORCNT
#  define WAIT_TENSORCNT(n) asm volatile("s_wait_tensorcnt %0" ::"i"(n))
#endif

// ---------------------------------------------------------------------------
// 1) proposal_lines for s in {-1,0,1}
// ---------------------------------------------------------------------------
__global__ void lines_kernel(const float* __restrict__ md,
                             const float* __restrict__ dis,
                             const float* __restrict__ res,
                             float* __restrict__ lines) {
  int i = blockIdx.x * blockDim.x + threadIdx.x;
  if (i >= NLINES) return;
  const int HW = H_ * W_;
  int s_idx = i / HW;
  int pix   = i % HW;
  float s  = (float)(s_idx - 1);
  float y0 = (float)(pix / W_);
  float x0 = (float)(pix % W_);
  const float PI = 3.14159265358979323846f;

  float md0 = md[pix];
  float md1 = md[HW + pix];
  float md2 = md[2 * HW + pix];

  float md_ = (md0 - 0.5f) * PI * 2.0f;
  float st_ = md1 * PI * 0.5f;
  float ed_ = -md2 * PI * 0.5f;

  float cs_md = __cosf(md_), ss_md = __sinf(md_);
  float cs_st = fmaxf(__cosf(st_), 0.001f);
  float ss_st = fmaxf(__sinf(st_), 0.001f);
  float cs_ed = fmaxf(__cosf(ed_), 0.001f);
  float ss_ed = fminf(__sinf(ed_), -0.001f);

  float y_st = ss_st / cs_st;
  float y_ed = ss_ed / cs_ed;
  float d = (dis[pix] + s * res[pix]) * 5.0f;

  float x_st = (cs_md - ss_md * y_st) * d;
  float y_s  = (ss_md + cs_md * y_st) * d;
  float x_ed = (cs_md - ss_md * y_ed) * d;
  float y_e  = (ss_md + cs_md * y_ed) * d;

  float4 r;
  r.x = fminf(fmaxf(x_st + x0, 0.0f), (float)(W_ - 1));
  r.y = fminf(fmaxf(y_s  + y0, 0.0f), (float)(H_ - 1));
  r.z = fminf(fmaxf(x_ed + x0, 0.0f), (float)(W_ - 1));
  r.w = fminf(fmaxf(y_e  + y0, 0.0f), (float)(H_ - 1));
  ((float4*)lines)[i] = r;
}

// ---------------------------------------------------------------------------
// 2) line pooling: wave32 per line, lane == sample point t.
// ---------------------------------------------------------------------------
__global__ __launch_bounds__(256)
void pool_kernel(const float* __restrict__ feats,
                 const float* __restrict__ lines,
                 _Float16* __restrict__ xout) {
  int lane = threadIdx.x & 31;
  int line = blockIdx.x * (blockDim.x >> 5) + (threadIdx.x >> 5);
  if (line >= NLINES) return;

  float4 L = ((const float4*)lines)[line];
  float t  = (float)lane * (1.0f / 31.0f);
  float px = L.x * t + L.z * (1.0f - t) - 0.5f;
  float py = L.y * t + L.w * (1.0f - t) - 0.5f;

  float px0 = fminf(fmaxf(floorf(px), 0.0f), 127.0f);
  float py0 = fminf(fmaxf(floorf(py), 0.0f), 127.0f);
  float px1 = fminf(px0 + 1.0f, 127.0f);
  float py1 = fminf(py0 + 1.0f, 127.0f);
  int ix0 = (int)px0, iy0 = (int)py0, ix1 = (int)px1, iy1 = (int)py1;

  float w00 = (py1 - py) * (px1 - px);
  float w10 = (py - py0) * (px1 - px);
  float w01 = (py1 - py) * (px - px0);
  float w11 = (py - py0) * (px - px0);

  int o00 = iy0 * W_ + ix0, o10 = iy1 * W_ + ix0;
  int o01 = iy0 * W_ + ix1, o11 = iy1 * W_ + ix1;

  int  p  = lane >> 2;
  bool wr = (lane & 3) == 0;
  _Float16* orow = xout + (size_t)line * DFEAT;

#pragma unroll 4
  for (int c = 0; c < C_; ++c) {
    const float* f = feats + c * (H_ * W_);
    float v = f[o00] * w00 + f[o10] * w10 + f[o01] * w01 + f[o11] * w11;
    float v2 = fmaxf(v, __shfl_xor(v, 1, 32));
    float v4 = fmaxf(v2, __shfl_xor(v2, 2, 32));
    if (wr) orow[c * NPTS1 + p] = (_Float16)v4;
  }
}

// ---------------------------------------------------------------------------
// f32 -> f16 weight conversion
// ---------------------------------------------------------------------------
__global__ void cvt_f32_f16(const float* __restrict__ src,
                            _Float16* __restrict__ dst, int n) {
  int i = blockIdx.x * blockDim.x + threadIdx.x;
  if (i < n) dst[i] = (_Float16)src[i];
}

// ---------------------------------------------------------------------------
// TDM: issue a 2D tile load (tile_d0 x tile_d1 elements of 2 bytes) from a
// row-major tensor (dim0 = fastest, stride0 elements between rows) into
// contiguous LDS at lds_off. Descriptor layout per CDNA5 ISA 8.3/8.4.
// ---------------------------------------------------------------------------
#if USE_TDM
__device__ __forceinline__ void tdm_load_2d(const _Float16* gptr, unsigned lds_off,
                                            unsigned tensor_d0, unsigned tensor_d1,
                                            unsigned long long stride0,
                                            unsigned tile_d0, unsigned tile_d1) {
  unsigned long long ga = (unsigned long long)gptr;
  u32x4 g0;
  g0[0] = 1u;                                             // count=1 (valid)
  g0[1] = lds_off;                                        // lds_addr
  g0[2] = (unsigned)(ga & 0xffffffffu);                   // global_addr[31:0]
  g0[3] = (unsigned)((ga >> 32) & 0x01ffffffu) | (2u << 30); // addr[56:32] | type=2
  i32x8 g1;
  g1[0] = (int)(1u << 16);                                // data_size=1 (2 bytes)
  g1[1] = (int)((tensor_d0 & 0xffffu) << 16);             // tensor_dim0[15:0]
  g1[2] = (int)((tensor_d0 >> 16) | ((tensor_d1 & 0xffffu) << 16));
  g1[3] = (int)((tensor_d1 >> 16) | (tile_d0 << 16));     // tile_dim0
  g1[4] = (int)tile_d1;                                   // tile_dim1 | tile_dim2=0
  g1[5] = (int)(unsigned)(stride0 & 0xffffffffu);         // dim0_stride[31:0]
  g1[6] = (int)(unsigned)((stride0 >> 32) & 0xffffu);     // dim0_stride[47:32]
  g1[7] = 0;
  i32x4 z = {0, 0, 0, 0};
#if __clang_major__ >= 23
  i32x8 z8 = {0, 0, 0, 0, 0, 0, 0, 0};
  __builtin_amdgcn_tensor_load_to_lds(g0, g1, z, z, z8, 0);
#else
  __builtin_amdgcn_tensor_load_to_lds(g0, g1, z, z, 0);
#endif
}
#endif

__device__ __forceinline__ unsigned lds_byte_off(const void* p) {
  // LDS aperture keeps the LDS byte offset in the low 32 bits of the flat addr.
  return (unsigned)(unsigned long long)(uintptr_t)p;
}

// ---------------------------------------------------------------------------
// 3) WMMA GEMM, TDM-staged, double-buffered.
//    Cout[m,n] = act(sum_k A[m,k]*B[n,k] + bias[n]) as f16.
//    Block = 256 thr (8 waves), tile 128x128, BK = 64, 2 LDS stages (64 KB).
//    Wave 0 drives the Tensor Data Mover; stage s+1 is in flight while all
//    waves run 16 x v_wmma_f32_16x16x32_f16 on stage s.
// ---------------------------------------------------------------------------
__global__ __launch_bounds__(256)
void gemm_wmma(const _Float16* __restrict__ A,
               const _Float16* __restrict__ B,
               const float* __restrict__ bias,
               _Float16* __restrict__ Cout,
               int M, int N, int K, int relu) {
  __shared__ __align__(32) _Float16 As[2][128][BK];
  __shared__ __align__(32) _Float16 Bs[2][128][BK];

  int nb = N >> 7;
  int m0 = (blockIdx.x / nb) << 7;
  int n0 = (blockIdx.x % nb) << 7;

  int tid  = threadIdx.x;
  int lane = tid & 31;
  int wid  = tid >> 5;
  int wm   = wid & 3;       // wave M index -> 32-row slab
  int wn   = wid >> 2;      // wave N index -> 64-col slab
  int l16  = lane & 15;
  int half = lane >> 4;

  v8f acc[2][4] = {};

  // ---- stage 0 prefetch ----
#if USE_TDM
  if (wid == 0) {
    tdm_load_2d(A + (size_t)m0 * K, lds_byte_off(&As[0][0][0]),
                (unsigned)K, (unsigned)M, (unsigned long long)K, BK, 128);
    tdm_load_2d(B + (size_t)n0 * K, lds_byte_off(&Bs[0][0][0]),
                (unsigned)K, (unsigned)N, (unsigned long long)K, BK, 128);
  }
#else
#pragma unroll
  for (int j = 0; j < 4; ++j) {
    int chunk = tid + 256 * j;          // 1024 chunks of 8 halves per tile
    int row   = chunk >> 3;
    int k8    = (chunk & 7) << 3;
    *(v8h*)&As[0][row][k8] = *(const v8h*)&A[(size_t)(m0 + row) * K + k8];
    *(v8h*)&Bs[0][row][k8] = *(const v8h*)&B[(size_t)(n0 + row) * K + k8];
  }
#endif

  int stage = 0;
  for (int kb = 0; kb < K; kb += BK, stage ^= 1) {
    int nxt = stage ^ 1;
#if USE_TDM
    if (wid == 0) {
      if (kb + BK < K) {
        tdm_load_2d(A + (size_t)m0 * K + kb + BK, lds_byte_off(&As[nxt][0][0]),
                    (unsigned)K, (unsigned)M, (unsigned long long)K, BK, 128);
        tdm_load_2d(B + (size_t)n0 * K + kb + BK, lds_byte_off(&Bs[nxt][0][0]),
                    (unsigned)K, (unsigned)N, (unsigned long long)K, BK, 128);
        WAIT_TENSORCNT(2);   // stage `stage` complete; next 2 still in flight
      } else {
        WAIT_TENSORCNT(0);
      }
    }
#else
    if (kb + BK < K) {
#pragma unroll
      for (int j = 0; j < 4; ++j) {
        int chunk = tid + 256 * j;
        int row   = chunk >> 3;
        int k8    = (chunk & 7) << 3;
        *(v8h*)&As[nxt][row][k8] = *(const v8h*)&A[(size_t)(m0 + row) * K + kb + BK + k8];
        *(v8h*)&Bs[nxt][row][k8] = *(const v8h*)&B[(size_t)(n0 + row) * K + kb + BK + k8];
      }
    }
#endif
    __syncthreads();

    // ---- consume stage: 2 k-substeps x (2x4) wmma tiles ----
#pragma unroll
    for (int ks = 0; ks < BK; ks += 32) {
      v16h af[2];
#pragma unroll
      for (int i = 0; i < 2; ++i) {
        int r = wm * 32 + i * 16 + l16;
        af[i] = *(const v16h*)&As[stage][r][ks + half * 16];      // 16 contiguous K
      }
      v16h bf[4];
#pragma unroll
      for (int j = 0; j < 4; ++j) {
        int cidx = wn * 64 + j * 16 + l16;
        bf[j] = *(const v16h*)&Bs[stage][cidx][ks + half * 16];   // 16 contiguous K
      }
#pragma unroll
      for (int i = 0; i < 2; ++i)
#pragma unroll
        for (int j = 0; j < 4; ++j)
          acc[i][j] = __builtin_amdgcn_wmma_f32_16x16x32_f16(
              false, af[i], false, bf[j], (short)0, acc[i][j], false, false);
    }
    __syncthreads();
  }

  // ---- epilogue: bias + ReLU + f16 store ----
#pragma unroll
  for (int i = 0; i < 2; ++i)
#pragma unroll
    for (int j = 0; j < 4; ++j) {
      int n = n0 + wn * 64 + j * 16 + l16;
      float bv = bias[n];
#pragma unroll
      for (int r = 0; r < 8; ++r) {
        int m = m0 + wm * 32 + i * 16 + r + half * 8;
        float v = acc[i][j][r] + bv;
        if (relu) v = fmaxf(v, 0.0f);
        Cout[(size_t)m * N + n] = (_Float16)v;
      }
    }
}

// ---------------------------------------------------------------------------
// 4) final 1024 -> 1 projection: wave per row, shuffle reduction
// ---------------------------------------------------------------------------
__global__ __launch_bounds__(256)
void gemm3_kernel(const _Float16* __restrict__ A,
                  const _Float16* __restrict__ w,
                  const float* __restrict__ b3,
                  float* __restrict__ out) {
  int lane = threadIdx.x & 31;
  int m = blockIdx.x * (blockDim.x >> 5) + (threadIdx.x >> 5);
  if (m >= NLINES) return;
  const _Float16* row = A + (size_t)m * HID;
  float s = 0.0f;
  for (int k = lane; k < HID; k += 32)
    s += (float)row[k] * (float)w[k];
#pragma unroll
  for (int off = 16; off; off >>= 1) s += __shfl_xor(s, off, 32);
  if (lane == 0) out[m] = s + b3[0];
}

// ---------------------------------------------------------------------------
// 5) NMS (3x3 SAME max) and iterative top-50 junction extraction
// ---------------------------------------------------------------------------
__global__ void nms_kernel(const float* __restrict__ jloc,
                           float* __restrict__ out) {
  int i = blockIdx.x * blockDim.x + threadIdx.x;
  if (i >= H_ * W_) return;
  int y = i / W_, x = i % W_;
  float a = jloc[i];
  float mx = -__builtin_inff();
  for (int dy = -1; dy <= 1; ++dy)
    for (int dx = -1; dx <= 1; ++dx) {
      int yy = y + dy, xx = x + dx;
      if (yy >= 0 && yy < H_ && xx >= 0 && xx < W_)
        mx = fmaxf(mx, jloc[yy * W_ + xx]);
    }
  out[i] = (a == mx) ? a : 0.0f;
}

__global__ void topk_kernel(float* __restrict__ sc,
                            const float* __restrict__ joff,
                            float* __restrict__ juncs,
                            float* __restrict__ scores) {
  __shared__ float sval[256];
  __shared__ int   sidx[256];
  int tid = threadIdx.x;
  for (int it = 0; it < TOPK; ++it) {
    float best = -__builtin_inff();
    int   bidx = H_ * W_;
    for (int j = tid; j < H_ * W_; j += 256) {
      float v = sc[j];
      if (v > best) { best = v; bidx = j; }
    }
    sval[tid] = best; sidx[tid] = bidx;
    __syncthreads();
    for (int s = 128; s > 0; s >>= 1) {
      if (tid < s) {
        if (sval[tid + s] > sval[tid] ||
            (sval[tid + s] == sval[tid] && sidx[tid + s] < sidx[tid])) {
          sval[tid] = sval[tid + s];
          sidx[tid] = sidx[tid + s];
        }
      }
      __syncthreads();
    }
    if (tid == 0) {
      int idx = sidx[0];
      juncs[2 * it]     = (float)(idx % W_) + joff[idx] + 0.5f;
      juncs[2 * it + 1] = (float)(idx / W_) + joff[H_ * W_ + idx] + 0.5f;
      scores[it] = sval[0];
      sc[idx] = -__builtin_inff();
    }
    __syncthreads();
  }
}

// ---------------------------------------------------------------------------
// launcher
// ---------------------------------------------------------------------------
extern "C" void kernel_launch(void* const* d_in, const int* in_sizes, int n_in,
                              void* d_out, int out_size, void* d_ws, size_t ws_size,
                              hipStream_t stream) {
  const float* md    = (const float*)d_in[0];
  const float* dis   = (const float*)d_in[1];
  const float* res   = (const float*)d_in[2];
  const float* jloc  = (const float*)d_in[3];
  const float* joff  = (const float*)d_in[4];
  const float* feats = (const float*)d_in[5];
  const float* W1    = (const float*)d_in[6];
  const float* b1    = (const float*)d_in[7];
  const float* W2    = (const float*)d_in[8];
  const float* b2    = (const float*)d_in[9];
  const float* W3    = (const float*)d_in[10];
  const float* b3    = (const float*)d_in[11];
  float* out = (float*)d_out;

  char* ws = (char*)d_ws;
  size_t off = 0;
  auto carve = [&](size_t bytes) -> void* {
    void* p = ws + off;
    off = (off + bytes + 255) & ~(size_t)255;
    return p;
  };
  float*     lines  = (float*)carve((size_t)NLINES * 4 * sizeof(float));
  _Float16*  xbuf   = (_Float16*)carve((size_t)NLINES * DFEAT * sizeof(_Float16));
  _Float16*  hbuf   = (_Float16*)carve((size_t)NLINES * HID * sizeof(_Float16));
  _Float16*  W1h    = (_Float16*)carve((size_t)HID * DFEAT * sizeof(_Float16));
  _Float16*  W2h    = (_Float16*)carve((size_t)HID * HID * sizeof(_Float16));
  _Float16*  W3h    = (_Float16*)carve((size_t)HID * sizeof(_Float16));
  float*     nmsbuf = (float*)carve((size_t)H_ * W_ * sizeof(float));

  lines_kernel<<<(NLINES + 255) / 256, 256, 0, stream>>>(md, dis, res, lines);
  cvt_f32_f16<<<(HID * DFEAT + 255) / 256, 256, 0, stream>>>(W1, W1h, HID * DFEAT);
  cvt_f32_f16<<<(HID * HID + 255) / 256, 256, 0, stream>>>(W2, W2h, HID * HID);
  cvt_f32_f16<<<(HID + 255) / 256, 256, 0, stream>>>(W3, W3h, HID);

  pool_kernel<<<NLINES / 8, 256, 0, stream>>>(feats, lines, xbuf);

  int gblocks = (NLINES / 128) * (DFEAT / 128);   // 384 * 8 = 3072
  gemm_wmma<<<gblocks, 256, 0, stream>>>(xbuf, W1h, b1, hbuf, NLINES, HID, DFEAT, 1);
  gemm_wmma<<<gblocks, 256, 0, stream>>>(hbuf, W2h, b2, xbuf, NLINES, HID, HID, 1);
  gemm3_kernel<<<NLINES / 8, 256, 0, stream>>>(xbuf, W3h, b3, out);

  nms_kernel<<<(H_ * W_ + 255) / 256, 256, 0, stream>>>(jloc, nmsbuf);
  topk_kernel<<<1, 256, 0, stream>>>(nmsbuf, joff, out + NLINES, out + NLINES + 2 * TOPK);
}
